// VectorQuantizer_3169685864512
// MI455X (gfx1250) — compile-verified, matrix operand order checked by reading, not run on hardware
//
#include <hip/hip_runtime.h>
#include <hip/hip_bf16.h>

typedef __attribute__((ext_vector_type(16))) __bf16 v16bf;
typedef __attribute__((ext_vector_type(8)))  __bf16 v8bf;
typedef __attribute__((ext_vector_type(8)))  float  v8f;
typedef __attribute__((ext_vector_type(4)))  float  v4f;
typedef __attribute__((ext_vector_type(4)))  int    v4i;

#define VQ_D      256
#define VQ_K      4096
#define VQ_NROWS  65536            /* B*S = 16*4096 */
#define VQ_MTILE  128
#define VQ_WAVES  8
#define VQ_STAGE  32               /* embeddings per pipeline stage */
#define VQ_NSTG   (VQ_K / VQ_STAGE)
#define LDS_PITCH 264              /* 256 + 8 pad bf16 -> 528B rows, conflict-free */

// ---------------------------------------------------------------------------
// Async global->LDS copy (gfx1250). Signature per hipcc diagnostic:
//   (v4i addrspace(1)*, v4i addrspace(3)*, imm offset, imm cpol)
// Falls back to a synchronous reg copy if the builtin is unavailable.
// ---------------------------------------------------------------------------
#if defined(__has_builtin)
#if __has_builtin(__builtin_amdgcn_global_load_async_to_lds_b128)
#define VQ_HAVE_ASYNC 1
#endif
#endif

typedef __attribute__((address_space(1))) v4i gv4i;
typedef __attribute__((address_space(3))) v4i lv4i;

#if defined(VQ_HAVE_ASYNC)
__device__ __forceinline__ void vq_cp16(const void* g, void* l) {
  __builtin_amdgcn_global_load_async_to_lds_b128(
      (gv4i*)g, (lv4i*)l, /*offset=*/0, /*cpol=*/0);
}
#if __has_builtin(__builtin_amdgcn_s_wait_asynccnt)
#define VQ_WAIT_ASYNC(n) __builtin_amdgcn_s_wait_asynccnt(n)
#else
#define VQ_WAIT_ASYNC(n) asm volatile("s_wait_asynccnt %0" ::"i"(n) : "memory")
#endif
#else
__device__ __forceinline__ void vq_cp16(const void* g, void* l) {
  *(v8bf*)l = *(const v8bf*)g;
}
#define VQ_WAIT_ASYNC(n) ((void)0)
#endif

// ---------------------------------------------------------------------------
// Prep: split codebook f32 -> (hi, lo) bf16 and compute |e_k|^2.
// ---------------------------------------------------------------------------
__global__ void vq_prep(const float* __restrict__ E,
                        __bf16* __restrict__ Ehi,
                        __bf16* __restrict__ Elo,
                        float* __restrict__ norms) {
  const int k = blockIdx.x;
  const int d = threadIdx.x;
  float e = E[(size_t)k * VQ_D + d];
  __bf16 h = (__bf16)e;
  Ehi[(size_t)k * VQ_D + d] = h;
  Elo[(size_t)k * VQ_D + d] = (__bf16)(e - (float)h);

  __shared__ float red[VQ_D];
  red[d] = e * e;
  __syncthreads();
  #pragma unroll
  for (int s = VQ_D / 2; s > 0; s >>= 1) {
    if (d < s) red[d] += red[d + s];
    __syncthreads();
  }
  if (d == 0) norms[k] = red[0];
}

// ---------------------------------------------------------------------------
// Main kernel.
// ---------------------------------------------------------------------------
__global__ void __launch_bounds__(VQ_WAVES * 32) vq_main(
    const float* __restrict__ x,
    const float* __restrict__ E,
    const __bf16* __restrict__ Ehi,
    const __bf16* __restrict__ Elo,
    const float* __restrict__ norms,
    float* __restrict__ quant,
    int* __restrict__ prop) {
  const int lane = threadIdx.x & 31;
  const int wave = threadIdx.x >> 5;
  const int m0   = blockIdx.x * VQ_MTILE + wave * 16;
  const int lr   = lane & 15;        // A: row-in-tile; B: column-in-tile
  const int sel  = (lane >> 4) & 1;  // 16-bit fragment half select

  // [buf][term(hi/lo)][row * LDS_PITCH]
  __shared__ __align__(16) __bf16 sb[2][2][VQ_STAGE * LDS_PITCH];

  // ---- A fragments: this wave's 16 x-rows, full D, split f32 -> hi/lo bf16.
  v16bf a_hi[8], a_lo[8];
  {
    const float* xr = x + (size_t)(m0 + lr) * VQ_D;
    #pragma unroll
    for (int c = 0; c < 8; ++c) {
      const int b0 = c * 32 + sel * 8;
      #pragma unroll
      for (int i = 0; i < 8; ++i) {
        float f0 = xr[b0 + i];
        __bf16 h0 = (__bf16)f0;
        a_hi[c][i] = h0;
        a_lo[c][i] = (__bf16)(f0 - (float)h0);
        float f1 = xr[b0 + 16 + i];
        __bf16 h1 = (__bf16)f1;
        a_hi[c][8 + i] = h1;
        a_lo[c][8 + i] = (__bf16)(f1 - (float)h1);
      }
    }
  }

  float bmin[8];
  int   bidx[8];
  #pragma unroll
  for (int v = 0; v < 8; ++v) { bmin[v] = __builtin_inff(); bidx[v] = 0; }

  union V16 { v16bf v; v8bf h[2]; };

  // Stage copier: wave w copies rows {w, w+8, w+16, w+24}; lane = 16B chunk.
  // Per wave: 8 async b128 ops per stage (4 hi + 4 lo) -> ASYNCcnt budget 8.
  #define VQ_ISSUE_STAGE(stg)                                                  \
    {                                                                          \
      const int _n0 = (stg) * VQ_STAGE;                                        \
      const int _bf = (stg) & 1;                                               \
      _Pragma("unroll")                                                        \
      for (int _i = 0; _i < 4; ++_i) {                                         \
        const int _row = wave + _i * 8;                                        \
        vq_cp16(Ehi + (size_t)(_n0 + _row) * VQ_D + lane * 8,                  \
                &sb[_bf][0][_row * LDS_PITCH + lane * 8]);                     \
      }                                                                        \
      _Pragma("unroll")                                                        \
      for (int _i = 0; _i < 4; ++_i) {                                         \
        const int _row = wave + _i * 8;                                        \
        vq_cp16(Elo + (size_t)(_n0 + _row) * VQ_D + lane * 8,                  \
                &sb[_bf][1][_row * LDS_PITCH + lane * 8]);                     \
      }                                                                        \
    }

  VQ_ISSUE_STAGE(0);

  for (int j = 0; j < VQ_NSTG; ++j) {
    if (j + 1 < VQ_NSTG) {
      VQ_ISSUE_STAGE(j + 1);
      VQ_WAIT_ASYNC(8);   // drain stage j (only the 8 newest may remain)
    } else {
      VQ_WAIT_ASYNC(0);
    }
    __syncthreads();      // publish stage j LDS to all waves

    const __bf16* shi = &sb[j & 1][0][0];
    const __bf16* slo = &sb[j & 1][1][0];
    const __bf16* r0h = shi + lr * LDS_PITCH;
    const __bf16* r1h = shi + (16 + lr) * LDS_PITCH;
    const __bf16* r0l = slo + lr * LDS_PITCH;
    const __bf16* r1l = slo + (16 + lr) * LDS_PITCH;

    v8f acc0 = {}, acc1 = {};
    #pragma unroll
    for (int c = 0; c < 8; ++c) {
      const int b0 = c * 32 + sel * 8;
      V16 bh0, bl0, bh1, bl1;
      bh0.h[0] = *(const v8bf*)(r0h + b0); bh0.h[1] = *(const v8bf*)(r0h + b0 + 16);
      bh1.h[0] = *(const v8bf*)(r1h + b0); bh1.h[1] = *(const v8bf*)(r1h + b0 + 16);
      bl0.h[0] = *(const v8bf*)(r0l + b0); bl0.h[1] = *(const v8bf*)(r0l + b0 + 16);
      bl1.h[0] = *(const v8bf*)(r1l + b0); bl1.h[1] = *(const v8bf*)(r1l + b0 + 16);
      // Interleave the two independent accumulator chains.
      acc0 = __builtin_amdgcn_wmma_f32_16x16x32_bf16(false, a_hi[c], false, bh0.v, (short)0, acc0, false, false);
      acc1 = __builtin_amdgcn_wmma_f32_16x16x32_bf16(false, a_hi[c], false, bh1.v, (short)0, acc1, false, false);
      acc0 = __builtin_amdgcn_wmma_f32_16x16x32_bf16(false, a_lo[c], false, bh0.v, (short)0, acc0, false, false);
      acc1 = __builtin_amdgcn_wmma_f32_16x16x32_bf16(false, a_lo[c], false, bh1.v, (short)0, acc1, false, false);
      acc0 = __builtin_amdgcn_wmma_f32_16x16x32_bf16(false, a_hi[c], false, bl0.v, (short)0, acc0, false, false);
      acc1 = __builtin_amdgcn_wmma_f32_16x16x32_bf16(false, a_hi[c], false, bl1.v, (short)0, acc1, false, false);
    }

    const int n0 = j * VQ_STAGE + lr;
    const float nn0 = norms[n0];
    const float nn1 = norms[n0 + 16];
    // C layout: lane -> N, VGPR v -> M = v + (lane<16 ? 0 : 8)
    #pragma unroll
    for (int v = 0; v < 8; ++v) {
      float s0 = nn0 - 2.0f * acc0[v];
      if (s0 < bmin[v]) { bmin[v] = s0; bidx[v] = n0; }
      float s1 = nn1 - 2.0f * acc1[v];
      if (s1 < bmin[v]) { bmin[v] = s1; bidx[v] = n0 + 16; }
    }

    __syncthreads();      // all waves done with buf (j&1) before its reuse
  }

  // Reduce argmin across the 16 lanes of each half (xor masks stay in-half).
  #pragma unroll
  for (int m = 1; m <= 8; m <<= 1) {
    #pragma unroll
    for (int v = 0; v < 8; ++v) {
      float os = __shfl_xor(bmin[v], m, 32);
      int   oi = __shfl_xor(bidx[v], m, 32);
      if (os < bmin[v] || (os == bmin[v] && oi < bidx[v])) {
        bmin[v] = os; bidx[v] = oi;
      }
    }
  }

  // Write indices + gather winning codebook rows (f32, coalesced).
  const int half8 = (lane >> 4) * 8;
  #pragma unroll
  for (int v = 0; v < 8; ++v) {
    const int row = m0 + half8 + v;
    const int bi  = bidx[v];
    if (lr == 0) prop[row] = bi;
    const v4f* src = (const v4f*)(E + (size_t)bi * VQ_D);
    v4f*       dst = (v4f*)(quant + (size_t)row * VQ_D);
    #pragma unroll
    for (int j2 = 0; j2 < 4; ++j2)
      dst[lr + 16 * j2] = src[lr + 16 * j2];
  }
  #undef VQ_ISSUE_STAGE
}

// ---------------------------------------------------------------------------
extern "C" void kernel_launch(void* const* d_in, const int* in_sizes, int n_in,
                              void* d_out, int out_size, void* d_ws, size_t ws_size,
                              hipStream_t stream) {
  const float* x = (const float*)d_in[0];        // [65536, 256] f32
  const float* E = (const float*)d_in[1];        // [4096, 256] f32

  char* ws = (char*)d_ws;
  __bf16* Ehi   = (__bf16*)ws;                                   // 2 MB
  __bf16* Elo   = (__bf16*)(ws + (size_t)VQ_K * VQ_D * 2);       // 2 MB
  float*  norms = (float*)(ws + (size_t)VQ_K * VQ_D * 4);        // 16 KB

  float* quant = (float*)d_out;                                  // [65536,256]
  int*   prop  = (int*)((float*)d_out + (size_t)VQ_NROWS * VQ_D);// [65536]

  vq_prep<<<VQ_K, VQ_D, 0, stream>>>(E, Ehi, Elo, norms);
  vq_main<<<VQ_NROWS / VQ_MTILE, VQ_WAVES * 32, 0, stream>>>(
      x, E, Ehi, Elo, norms, quant, prop);
}
